// MultiHeadedSelfAttention_64518998720829
// MI455X (gfx1250) — matrix operand order repaired and checked by reading, not compiled, not run on hardware
//
#include <hip/hip_runtime.h>
#include <hip/hip_bf16.h>

typedef _Float16 v16h __attribute__((ext_vector_type(16)));
typedef _Float16 v8h  __attribute__((ext_vector_type(8)));
typedef float    v8f  __attribute__((ext_vector_type(8)));

#define DCH   768
#define NHEAD 12
#define HDIM  64
#define HW    32
#define TOK   1024          // H*W
#define LDP   772           // padded LDS row stride in halves (conflict-free)
#define WSZ   589824        // 768*768

// ---------------------------------------------------------------------------
// Kernel 1: per-token LayerNorm over D. One block per (b,h): 32 tokens (w).
// Writes xn as f16, (B*1024, 768) row-major, coalesced.
// ---------------------------------------------------------------------------
__global__ void __launch_bounds__(256) k_layernorm(
    const float* __restrict__ x, const float* __restrict__ gamma,
    const float* __restrict__ beta, _Float16* __restrict__ xnh)
{
  __shared__ _Float16 tile[32 * LDP];
  __shared__ float redS[8][32], redQ[8][32];
  __shared__ float muS[32], rsS[32];

  const int b = blockIdx.x >> 5;
  const int h = blockIdx.x & 31;
  const int w = threadIdx.x & 31;     // lane within wave (wave32)
  const int g = threadIdx.x >> 5;     // wave id 0..7, each owns 96 channels

  float s = 0.f, q = 0.f;
  const int d0 = g * 96;
  for (int dd = 0; dd < 96; ++dd) {
    int d = d0 + dd;
    float val = x[((size_t)(b * DCH + d) * HW + h) * HW + w]; // lanes over w: coalesced
    tile[w * LDP + d] = (_Float16)val;
    s += val; q += val * val;
  }
  redS[g][w] = s; redQ[g][w] = q;
  __syncthreads();
  if (g == 0) {
    float ts = 0.f, tq = 0.f;
    for (int gg = 0; gg < 8; ++gg) { ts += redS[gg][w]; tq += redQ[gg][w]; }
    float mu  = ts * (1.f / 768.f);
    float var = tq * (1.f / 768.f) - mu * mu;
    muS[w] = mu;
    rsS[w] = rsqrtf(var + 1e-6f);
  }
  __syncthreads();
  const float mu = muS[w], rs = rsS[w];
  for (int dd = 0; dd < 96; ++dd) {
    int d = d0 + dd;
    float val = (float)tile[w * LDP + d];
    float xn  = (val - mu) * rs * gamma[d] + beta[d];
    tile[w * LDP + d] = (_Float16)xn;
  }
  __syncthreads();
  // coalesced f16 write-out: lanes over d
  _Float16* dst = xnh + (size_t)(b * TOK + h * HW) * DCH;
  for (int t = 0; t < 32; ++t)
    for (int rep = 0; rep < 3; ++rep) {
      int d = rep * 256 + threadIdx.x;
      dst[(size_t)t * DCH + d] = tile[t * LDP + d];
    }
}

// ---------------------------------------------------------------------------
// Kernel 2: axial pooling of xn.
// xph row layout: rows [0,1024)  = y-sequences: row b*32+h = mean over w
//                 rows [1024,2048)= x-sequences: row 1024+b*32+w = mean over h
// ---------------------------------------------------------------------------
__global__ void k_pool(const _Float16* __restrict__ xnh, _Float16* __restrict__ xph)
{
  int idx  = blockIdx.x * 256 + threadIdx.x;      // 0 .. 2*1024*768-1 (exact grid)
  int dcol = idx % DCH;
  int row  = (idx / DCH) & (TOK - 1);             // b*32 + s
  int axis = idx / (TOK * DCH);                   // 0 = y, 1 = x
  int b = row >> 5, sidx = row & 31;
  size_t base; int stride;
  if (axis == 0) { base = ((size_t)b * TOK + sidx * HW) * DCH + dcol; stride = DCH;      }
  else           { base = ((size_t)b * TOK + sidx)      * DCH + dcol; stride = HW * DCH; }
  float acc = 0.f;
  for (int i = 0; i < 32; ++i) acc += (float)xnh[base + (size_t)i * stride];
  xph[idx] = (_Float16)(acc * (1.f / 32.f));
}

// ---------------------------------------------------------------------------
// Kernel 3: convert Wq/Wk/Wv to f16 (packed contiguously)
// ---------------------------------------------------------------------------
__global__ void k_packw(const float* __restrict__ Wq, const float* __restrict__ Wk,
                        const float* __restrict__ Wv, _Float16* __restrict__ Wh)
{
  int idx = blockIdx.x * 256 + threadIdx.x;       // 0 .. 3*589824-1 (exact)
  const float* src = (idx < WSZ) ? Wq : ((idx < 2 * WSZ) ? Wk : Wv);
  Wh[idx] = (_Float16)src[idx % WSZ];
}

// ---------------------------------------------------------------------------
// Kernel 4: fused QKV GEMM on pooled tokens via WMMA.
// y[t,j] = sum_k xph[t,k] * W[j,k] + bias[j]   (M=2048, N=768, K=768, 3 mats)
// One wave computes a 64x16 tile of all three outputs (A reuse x3, B x4).
// ---------------------------------------------------------------------------
__global__ void __launch_bounds__(256) k_qkv_gemm(
    const _Float16* __restrict__ xph, const _Float16* __restrict__ Wh,
    const float* __restrict__ bq, const float* __restrict__ bk,
    const float* __restrict__ bv, float* __restrict__ qkvp)
{
  const int lane = threadIdx.x & 31;
  const int gw   = blockIdx.x * 8 + (threadIdx.x >> 5);  // 1536 waves total
  const int nt   = gw % 48;                               // N tile (16 cols)
  const int mt   = gw / 48;                               // M tile (64 rows)
  const int mbase = mt * 64, jbase = nt * 16;
  const int r  = lane & 15;        // A: row-in-tile / B: column-in-tile
  const int kh = lane >> 4;        // half-lane select

  v8f acc[3][4] = {};

  for (int k0 = 0; k0 < DCH; k0 += 32) {
    v16h a[4];
#pragma unroll
    for (int i = 0; i < 4; ++i) {
      // A tile (16x32), contiguous k as in B: lanes 0-15 K={0..15}, lanes 16-31 K={16..31}
      const _Float16* pa = xph + (size_t)(mbase + i * 16 + r) * DCH + k0 + kh * 16;
      v8h lo = *(const v8h*)pa;
      v8h hi = *(const v8h*)(pa + 8);
      a[i] = __builtin_shufflevector(lo, hi, 0,1,2,3,4,5,6,7,8,9,10,11,12,13,14,15);
    }
#pragma unroll
    for (int v = 0; v < 3; ++v) {
      // B layout (32x16): lane = column n (mod 16), K chunk [kh*16, kh*16+16) contiguous
      const _Float16* pb = Wh + (size_t)v * WSZ + (size_t)(jbase + r) * DCH + k0 + kh * 16;
      v16h bt = *(const v16h*)pb;
#pragma unroll
      for (int i = 0; i < 4; ++i)
        acc[v][i] = __builtin_amdgcn_wmma_f32_16x16x32_f16(
            false, a[i], false, bt, (short)0, acc[v][i], false, false);
    }
  }

  const float* biases[3] = {bq, bk, bv};
#pragma unroll
  for (int v = 0; v < 3; ++v) {
    float bb = biases[v][jbase + r];                 // C/D layout: column n = lane&15
    float* outp = qkvp + (size_t)v * (2048 * DCH);
#pragma unroll
    for (int i = 0; i < 4; ++i)
#pragma unroll
      for (int e = 0; e < 8; ++e) {
        int row = mbase + i * 16 + kh * 8 + e;       // M = vgpr + (lane>>4)*8
        outp[(size_t)row * DCH + jbase + r] = acc[v][i][e] + bb;
      }
  }
}

// ---------------------------------------------------------------------------
// Kernel 5: axial attention (no softmax). One block per (axis, b, head).
// scores = q k^T / 8 ; out = scores v ; S=32, hd=64.
// ---------------------------------------------------------------------------
__global__ void __launch_bounds__(256) k_axial_attn(
    const float* __restrict__ qkvp, float* __restrict__ hout)
{
  __shared__ float sq[32][HDIM + 1], sk[32][HDIM + 1], sv[32][HDIM + 1];
  __shared__ float ss[32][32];
  const int nh    = blockIdx.x % NHEAD;
  const int tbase = (blockIdx.x / NHEAD) * 32;       // row base in 2048-row space
  const int tid   = threadIdx.x;

  const float* qp = qkvp;
  const float* kp = qkvp + (size_t)2048 * DCH;
  const float* vp = qkvp + (size_t)2 * 2048 * DCH;
  for (int i = 0; i < 8; ++i) {
    int idx = tid + i * 256;                         // 0..2047
    int s = idx >> 6, dd = idx & 63;
    size_t off = (size_t)(tbase + s) * DCH + nh * HDIM + dd;
    sq[s][dd] = qp[off]; sk[s][dd] = kp[off]; sv[s][dd] = vp[off];
  }
  __syncthreads();
  for (int i = 0; i < 4; ++i) {
    int idx = tid + i * 256;                         // 0..1023
    int qi = idx >> 5, ki = idx & 31;
    float a = 0.f;
    for (int dd = 0; dd < HDIM; ++dd) a += sq[qi][dd] * sk[ki][dd];
    ss[qi][ki] = a * 0.125f;                         // 1/sqrt(64)
  }
  __syncthreads();
  for (int i = 0; i < 8; ++i) {
    int idx = tid + i * 256;
    int s = idx >> 6, dd = idx & 63;
    float a = 0.f;
    for (int kk = 0; kk < 32; ++kk) a += ss[s][kk] * sv[kk][dd];
    hout[(size_t)(tbase + s) * DCH + nh * HDIM + dd] = a;
  }
}

// ---------------------------------------------------------------------------
// Kernel 6: out[b, h*32+w, d] = h_y[b,h,d] * h_x[b,w,d]  (float4 vectorized)
// ---------------------------------------------------------------------------
__global__ void k_outer(const float* __restrict__ hout, float* __restrict__ out)
{
  int idx = blockIdx.x * 256 + threadIdx.x;          // float4 index, exact grid
  int dv = idx % 192;
  int w  = (idx / 192) & 31;
  int h  = (idx / (192 * 32)) & 31;
  int b  = idx / (192 * 1024);
  const float4* hy = (const float4*)hout;                       // rows [0,1024)
  const float4* hx = (const float4*)(hout + (size_t)TOK * DCH); // rows [1024,2048)
  float4 a = hy[(size_t)(b * 32 + h) * 192 + dv];
  float4 c = hx[(size_t)(b * 32 + w) * 192 + dv];
  float4 o; o.x = a.x * c.x; o.y = a.y * c.y; o.z = a.z * c.z; o.w = a.w * c.w;
  ((float4*)out)[idx] = o;
}

// ---------------------------------------------------------------------------
extern "C" void kernel_launch(void* const* d_in, const int* in_sizes, int n_in,
                              void* d_out, int out_size, void* d_ws, size_t ws_size,
                              hipStream_t stream)
{
  const float* x     = (const float*)d_in[0];
  const float* Wq    = (const float*)d_in[1];
  const float* bq    = (const float*)d_in[2];
  const float* Wk    = (const float*)d_in[3];
  const float* bk    = (const float*)d_in[4];
  const float* Wv    = (const float*)d_in[5];
  const float* bv    = (const float*)d_in[6];
  const float* gamma = (const float*)d_in[7];
  const float* beta  = (const float*)d_in[8];
  float* out = (float*)d_out;

  // workspace layout (all offsets 256B-aligned); total ~82.2 MB
  char* ws = (char*)d_ws;
  _Float16* xnh = (_Float16*)(ws);              // 32768*768 f16 = 50,331,648 B
  _Float16* xph = (_Float16*)(ws + 50331648);   //  2048*768 f16 =  3,145,728 B
  _Float16* Wh  = (_Float16*)(ws + 53477376);   // 3*768*768 f16 =  3,538,944 B
  float*    qkvp= (float*)   (ws + 57016320);   // 3*2048*768 f32 = 18,874,368 B
  float*    hout= (float*)   (ws + 75890688);   // 2*1024*768 f32 =  6,291,456 B

  k_layernorm <<<1024,                 256, 0, stream>>>(x, gamma, beta, xnh);
  k_pool      <<<(2 * TOK * DCH) / 256,256, 0, stream>>>(xnh, xph);
  k_packw     <<<(3 * WSZ) / 256,      256, 0, stream>>>(Wq, Wk, Wv, Wh);
  k_qkv_gemm  <<<192,                  256, 0, stream>>>(xph, Wh, bq, bk, bv, qkvp);
  k_axial_attn<<<64 * NHEAD,           256, 0, stream>>>(qkvp, hout);
  k_outer     <<<(32 * TOK * 192) / 256,256, 0, stream>>>(hout, out);
}
